// RobustPLRNN_31645319037273
// MI455X (gfx1250) — matrix operand order repaired and checked by reading, not compiled, not run on hardware
//
#include <hip/hip_runtime.h>
#include <hip/hip_bf16.h>
#include <math.h>

// ---------------- CDNA5 WMMA types ----------------
typedef __attribute__((ext_vector_type(16))) _Float16 v16h;
typedef __attribute__((ext_vector_type(8)))  float    v8f;
typedef __attribute__((__vector_size__(16))) int      vi4;

union Frag16 { v16h v; _Float16 h[16]; uint4 q[2]; };
union Half8  { _Float16 h[8]; uint4 q; };
union Fl8    { float f[8]; float4 q[2]; };

// A-matrix 16x32 f16 (k order used here): per lane, one contiguous run of 16 halfs:
//   lanes 0-15 : K = 0..15   (row M = lane&15)
//   lanes16-31 : K = 16..31
// B-matrix 32x16 f16: lane n<16: col n, K=0..15 ; lane n+16: col n, K=16..31

__device__ __forceinline__ v8f wmma_f16(v16h a, v16h b, v8f c) {
    return __builtin_amdgcn_wmma_f32_16x16x32_f16(false, a, false, b, (short)0, c, false, false);
}

// load A-frag from row-major f16 [16 x stride] region (16B-aligned rows)
__device__ __forceinline__ void load_a(Frag16& a, const _Float16* base, int stride,
                                       int nlo, int lo) {
    const _Float16* p = base + nlo * stride + lo;
    a.q[0] = *(const uint4*)(p);
    a.q[1] = *(const uint4*)(p + 8);
}

// ---------------- optional CDNA5 async global->LDS path ----------------
#if defined(__has_builtin)
#if __has_builtin(__builtin_amdgcn_global_load_async_to_lds_b32) && \
    __has_builtin(__builtin_amdgcn_global_load_async_to_lds_b128) && \
    __has_builtin(__builtin_amdgcn_s_wait_asynccnt)
#define USE_ASYNC_LDS 1
#endif
#endif

__device__ __forceinline__ void copy_b32_to_lds(const float* g, float* l) {
#ifdef USE_ASYNC_LDS
    __builtin_amdgcn_global_load_async_to_lds_b32(
        (__attribute__((address_space(1))) int*)g,
        (__attribute__((address_space(3))) int*)l, 0, 0);
#else
    *l = *g;
#endif
}
__device__ __forceinline__ void copy_b128_to_lds(const _Float16* g, _Float16* l) {
#ifdef USE_ASYNC_LDS
    __builtin_amdgcn_global_load_async_to_lds_b128(
        (__attribute__((address_space(1))) vi4*)g,
        (__attribute__((address_space(3))) vi4*)l, 0, 0);
#else
    *(uint4*)l = *(const uint4*)g;
#endif
}
__device__ __forceinline__ void async_wait() {
#ifdef USE_ASYNC_LDS
    __builtin_amdgcn_s_wait_asynccnt(0);
#endif
}

// ---------------- problem constants ----------------
#define BB    512
#define CIN   12
#define LIN   5000
#define OC    32
#define KW1   50
#define S1    10
#define T1    496
#define K1TOT 600
#define NK1   19
#define KW2   25
#define S2    5
#define T2    95
#define K2TOT 800
#define NK2   25
#define NF    32
#define HID   64
#define LAT   32
#define EPS_  1e-5f

// ---------------- workspace layout (bytes) ----------------
#define OFF_A1    ((size_t)0)                  // 2*19*512 f16
#define OFF_A2    ((size_t)(64  * 1024))       // 2*25*512 f16
#define OFF_WHB   ((size_t)(192 * 1024))       // 4*512 f16
#define OFF_WRB   ((size_t)(200 * 1024))       // 4*512 f16
#define OFF_WOB   ((size_t)(208 * 1024))       // 2*512 f16
#define OFF_WIA   ((size_t)(212 * 1024))       // 2*512 f16
#define OFF_STATS ((size_t)(216 * 1024))       // 256 f32
#define OFF_Y1    ((size_t)(256 * 1024))       // [b][t][c] f16, 16.25 MB
#define OFF_Y2    ((size_t)(17 * 1024 * 1024)) // [t][b][c] f16, 3.1 MB
#define OFF_XT    ((size_t)(21 * 1024 * 1024)) // [t][b][lat] f32, 6.2 MB

// ================= prep kernels =================
__global__ void zero_f32(float* p, int n) {
    int i = blockIdx.x * blockDim.x + threadIdx.x;
    if (i < n) p[i] = 0.f;
}

__device__ __forceinline__ int amap_k(int lane, int j) {
    int lo = (lane < 16) ? 0 : 16;
    return lo + j; // contiguous: lo + j
}

// pack weights [32, ktot] into A-fragments: [mtile(2)][kc(nk)][lane(32)][j(16)]
__global__ void pack_a(const float* __restrict__ w, _Float16* __restrict__ dst,
                       int ktot, int nk) {
    int idx = blockIdx.x * blockDim.x + threadIdx.x;
    if (idx >= 2 * nk * 512) return;
    int j    = idx & 15;
    int lane = (idx >> 4) & 31;
    int kc   = (idx >> 9) % nk;
    int mt   = idx / (nk * 512);
    int m    = mt * 16 + (lane & 15);
    int K    = kc * 32 + amap_k(lane, j);
    float val = (K < ktot) ? w[m * ktot + K] : 0.f;
    dst[idx] = (_Float16)val;
}

// pack W [ntot, ktot] as B = W^T fragments: [kc][nt][lane][j]
__global__ void pack_b(const float* __restrict__ W, _Float16* __restrict__ dst,
                       int ktot, int ntot) {
    int ntiles = ntot / 16;
    int idx = blockIdx.x * blockDim.x + threadIdx.x;
    if (idx >= (ktot / 32) * ntiles * 512) return;
    int j    = idx & 15;
    int lane = (idx >> 4) & 31;
    int nt   = (idx >> 9) % ntiles;
    int kc   = idx / (ntiles * 512);
    int n = nt * 16 + (lane & 15);
    int k = kc * 32 + ((lane < 16) ? 0 : 16) + j;
    dst[idx] = (_Float16)W[n * ktot + k];
}

// ================= conv1: implicit GEMM =================
// y1t layout: [b][t][c] f16
__global__ void __launch_bounds__(64) conv1_kernel(
    const float* __restrict__ x, const _Float16* __restrict__ Apack,
    const float* __restrict__ bias, _Float16* __restrict__ y1t,
    float* __restrict__ sum, float* __restrict__ sumsq)
{
    __shared__ float    patchf[CIN * 200];       // raw x tile (f32)
    __shared__ _Float16 bfrag[NK1 * 512];        // B-fragments, frag-contiguous
    __shared__ float lsum[OC], lsq[OC];
    const int b  = blockIdx.y;
    const int t0 = blockIdx.x * 16;
    const int tid = threadIdx.x, wave = tid >> 5, lane = tid & 31;
    const int nlo = lane & 15, rofs = (lane < 16) ? 0 : 8;

    if (tid < OC) { lsum[tid] = 0.f; lsq[tid] = 0.f; }
    // stage raw input tile (async global->LDS when available)
    const float* xb = x + (size_t)b * (CIN * LIN) + t0 * S1;
    for (int i = tid; i < CIN * 200; i += 64)
        copy_b32_to_lds(xb + (i / 200) * LIN + (i % 200), patchf + i);
    async_wait();
    __syncthreads();

    // expand im2col B-fragments cooperatively (incremental addressing, no divides in j-loop)
    for (int r = tid; r < NK1 * 32; r += 64) {
        int kc = r >> 5, ln = r & 31, n = ln & 15;
        int K0 = kc * 32 + ((ln < 16) ? 0 : 16);
        int k  = K0 % KW1;
        int addr = (K0 / KW1) * 200 + n * S1 + k;
        Frag16 tmp;
#pragma unroll
        for (int j = 0; j < 16; ++j) {
            int K = K0 + j;
            tmp.h[j] = (K < K1TOT) ? (_Float16)patchf[addr] : (_Float16)0.f;
            ++k; ++addr;
            if (k == KW1) { k = 0; addr += 200 - KW1; }
        }
        uint4* dst = (uint4*)(bfrag + r * 16);
        dst[0] = tmp.q[0]; dst[1] = tmp.q[1];
    }
    __syncthreads();

    v8f acc = {};
    const _Float16* ap = Apack + (size_t)wave * (NK1 * 512) + lane * 16;
    __builtin_prefetch(ap, 0, 3);
#pragma unroll
    for (int kc = 0; kc < NK1; ++kc) {
        Frag16 a, bf;
        const uint4* ag = (const uint4*)(ap + kc * 512);
        a.q[0] = ag[0]; a.q[1] = ag[1];
        const uint4* bg = (const uint4*)(bfrag + (kc * 32 + lane) * 16);
        bf.q[0] = bg[0]; bf.q[1] = bg[1];
        acc = wmma_f16(a.v, bf.v, acc);
    }

    // epilogue: bias, vectorized store to [b][t][c], BN stats
    Half8 st;
#pragma unroll
    for (int v = 0; v < 8; ++v) {
        int oc = wave * 16 + rofs + v;
        float val = acc[v] + bias[oc];
        st.h[v] = (_Float16)val;
        atomicAdd(&lsum[oc], val);
        atomicAdd(&lsq[oc], val * val);
    }
    *(uint4*)(y1t + ((size_t)b * T1 + t0 + nlo) * OC + wave * 16 + rofs) = st.q;
    __syncthreads();
    if (tid < OC) { atomicAdd(&sum[tid], lsum[tid]); atomicAdd(&sumsq[tid], lsq[tid]); }
}

// ================= BN param solve =================
__global__ void bn_params(const float* __restrict__ sum, const float* __restrict__ sumsq,
                          const float* __restrict__ g, const float* __restrict__ bvec,
                          float* __restrict__ scale, float* __restrict__ shift, float invcnt)
{
    int c = threadIdx.x;
    if (c >= OC) return;
    float m   = sum[c] * invcnt;
    float var = sumsq[c] * invcnt - m * m;
    float sc  = g[c] * rsqrtf(var + EPS_);
    scale[c] = sc;
    shift[c] = bvec[c] - m * sc;
}

// ================= conv2: fused BN1+ReLU =================
// y2t layout: [t][b][c] f16
__global__ void __launch_bounds__(64) conv2_kernel(
    const _Float16* __restrict__ y1t, const _Float16* __restrict__ Apack,
    const float* __restrict__ bias, const float* __restrict__ bn1scale,
    const float* __restrict__ bn1shift, _Float16* __restrict__ y2t,
    float* __restrict__ sum2, float* __restrict__ sumsq2)
{
    __shared__ _Float16 patch2[100 * OC];        // [pos][c], raw y1t tile
    __shared__ _Float16 bfrag[NK2 * 512];
    __shared__ float ssc[OC], ssh[OC];
    __shared__ float lsum[OC], lsq[OC];
    const int b  = blockIdx.y;
    const int t0 = blockIdx.x * 16;
    const int tid = threadIdx.x, wave = tid >> 5, lane = tid & 31;
    const int nlo = lane & 15, rofs = (lane < 16) ? 0 : 8;

    if (tid < OC) {
        lsum[tid] = 0.f; lsq[tid] = 0.f;
        ssc[tid] = bn1scale[tid]; ssh[tid] = bn1shift[tid];
    }
    // stage raw [pos][c] tile, fully coalesced (one b128 per 8 halfs)
    const _Float16* ysrc = y1t + ((size_t)b * T1 + t0 * S2) * OC;
    for (int i8 = tid; i8 < (100 * OC) / 8; i8 += 64) {
        int pos = i8 >> 2;
        if (t0 * S2 + pos < T1) {
            copy_b128_to_lds(ysrc + i8 * 8, patch2 + i8 * 8);
        } else {
            uint4 z = {0, 0, 0, 0};
            *(uint4*)(patch2 + i8 * 8) = z;
        }
    }
    async_wait();
    __syncthreads();

    // expand B-fragments; BN1 + ReLU applied here (K2TOT = 800 = 25*32 exact)
    for (int r = tid; r < NK2 * 32; r += 64) {
        int kc = r >> 5, ln = r & 31, n = ln & 15;
        int K0 = kc * 32 + ((ln < 16) ? 0 : 16);
        int c  = K0 / KW2;
        int k  = K0 % KW2;
        int addr = (n * S2 + k) * OC + c;
        Frag16 tmp;
#pragma unroll
        for (int j = 0; j < 16; ++j) {
            float v = (float)patch2[addr] * ssc[c] + ssh[c];
            tmp.h[j] = (_Float16)(v > 0.f ? v : 0.f);
            ++k; addr += OC;
            if (k == KW2) { k = 0; ++c; addr += 1 - KW2 * OC; }
        }
        uint4* dst = (uint4*)(bfrag + r * 16);
        dst[0] = tmp.q[0]; dst[1] = tmp.q[1];
    }
    __syncthreads();

    v8f acc = {};
    const _Float16* ap = Apack + (size_t)wave * (NK2 * 512) + lane * 16;
    __builtin_prefetch(ap, 0, 3);
#pragma unroll
    for (int kc = 0; kc < NK2; ++kc) {
        Frag16 a, bf;
        const uint4* ag = (const uint4*)(ap + kc * 512);
        a.q[0] = ag[0]; a.q[1] = ag[1];
        const uint4* bg = (const uint4*)(bfrag + (kc * 32 + lane) * 16);
        bf.q[0] = bg[0]; bf.q[1] = bg[1];
        acc = wmma_f16(a.v, bf.v, acc);
    }

    const int t = t0 + nlo;
    const bool live = (t < T2);
    Half8 st;
    float ls[8], lq[8];
#pragma unroll
    for (int v = 0; v < 8; ++v) {
        int oc = wave * 16 + rofs + v;
        float val = acc[v] + bias[oc];
        st.h[v] = (_Float16)val;
        ls[v] = val; lq[v] = val * val;
    }
    if (live) {
        *(uint4*)(y2t + ((size_t)t * BB + b) * OC + wave * 16 + rofs) = st.q;
#pragma unroll
        for (int v = 0; v < 8; ++v) {
            int oc = wave * 16 + rofs + v;
            atomicAdd(&lsum[oc], ls[v]);
            atomicAdd(&lsq[oc], lq[v]);
        }
    }
    __syncthreads();
    if (tid < OC) { atomicAdd(&sum2[tid], lsum[tid]); atomicAdd(&sumsq2[tid], lsq[tid]); }
}

// ================= projection + LN1, WMMA version =================
// out[t,b,l] = LN_col( Wi @ relu(bn2(feat)) + bi ); 16 (t,b) columns per wave
__global__ void __launch_bounds__(64) proj_kernel(
    const _Float16* __restrict__ y2t, const float* __restrict__ bn2scale,
    const float* __restrict__ bn2shift, const _Float16* __restrict__ WiA,
    const float* __restrict__ bi, const float* __restrict__ ln1g,
    const float* __restrict__ ln1b, float* __restrict__ xt_all)
{
    const int tid = threadIdx.x, wave = tid >> 5, lane = tid & 31;
    const int nlo = lane & 15, rofs = (lane < 16) ? 0 : 8;
    const int clo = (lane < 16) ? 0 : 16;
    const int idx = blockIdx.x * 32 + wave * 16 + nlo;   // column = t*512+b

    // A-fragments of Wi (2 M-tiles), contiguous global loads
    Frag16 a0, a1;
    {
        const uint4* g0 = (const uint4*)(WiA + (size_t)lane * 16);
        const uint4* g1 = (const uint4*)(WiA + (size_t)(32 + lane) * 16);
        a0.q[0] = g0[0]; a0.q[1] = g0[1];
        a1.q[0] = g1[0]; a1.q[1] = g1[1];
    }
    // B-fragment: 16 contiguous halfs of this column's features, BN2+ReLU applied
    Frag16 bf;
    {
        const _Float16* src = y2t + (size_t)idx * OC + clo;
#pragma unroll
        for (int j = 0; j < 16; ++j) {
            int c = clo + j;
            float v = (float)src[j] * bn2scale[c] + bn2shift[c];
            bf.h[j] = (_Float16)(v > 0.f ? v : 0.f);
        }
    }
    v8f c0 = {}, c1 = {};
    c0 = wmma_f16(a0.v, bf.v, c0);
    c1 = wmma_f16(a1.v, bf.v, c1);

    // bias + column LayerNorm (column = M dim; one xor-16 shuffle completes the column)
    float v0[8], v1[8], s = 0.f, q = 0.f;
#pragma unroll
    for (int v = 0; v < 8; ++v) {
        v0[v] = c0[v] + bi[rofs + v];
        v1[v] = c1[v] + bi[16 + rofs + v];
        s += v0[v] + v1[v];
        q += v0[v] * v0[v] + v1[v] * v1[v];
    }
    s += __shfl_xor(s, 16, 32);
    q += __shfl_xor(q, 16, 32);
    float mean = s * (1.f / LAT);
    float var  = q * (1.f / LAT) - mean * mean;
    float r    = rsqrtf(var + EPS_);

    Fl8 o0, o1;
#pragma unroll
    for (int v = 0; v < 8; ++v) {
        o0.f[v] = (v0[v] - mean) * r * ln1g[rofs + v]      + ln1b[rofs + v];
        o1.f[v] = (v1[v] - mean) * r * ln1g[16 + rofs + v] + ln1b[16 + rofs + v];
    }
    float* dst = xt_all + (size_t)idx * LAT;
    *(float4*)(dst + rofs)          = o0.q[0];
    *(float4*)(dst + rofs + 4)      = o0.q[1];
    *(float4*)(dst + 16 + rofs)     = o1.q[0];
    *(float4*)(dst + 16 + rofs + 4) = o1.q[1];
}

// ================= RNN scan: 1 wave per 16 batch rows =================
__global__ void __launch_bounds__(32) rnn_kernel(
    const float* __restrict__ xt_all,       // [95][512][32]
    const _Float16* __restrict__ WhB, const float* __restrict__ bh,
    const float* __restrict__ ln2g, const float* __restrict__ ln2b,
    const _Float16* __restrict__ WrB, const float* __restrict__ br,
    const _Float16* __restrict__ WoB, const float* __restrict__ bo,
    float* __restrict__ out)
{
    __shared__ float    hbuf[16 * LAT];
    __shared__ _Float16 comb16[16 * LAT];
    __shared__ _Float16 hp16[16 * HID];
    const int lane = threadIdx.x;
    const int b0   = blockIdx.x * 16;
    const int nlo  = lane & 15;
    const int rofs = (lane < 16) ? 0 : 8;
    const int lo   = (lane < 16) ? 0 : 16;  // A-frag K base

    Frag16 whb[4], wrb[2][2], wob[2];
#pragma unroll
    for (int nt = 0; nt < 4; ++nt) {
        const uint4* g = (const uint4*)(WhB + (size_t)(nt * 32 + lane) * 16);
        whb[nt].q[0] = g[0]; whb[nt].q[1] = g[1];
    }
#pragma unroll
    for (int kc = 0; kc < 2; ++kc)
#pragma unroll
        for (int nt = 0; nt < 2; ++nt) {
            const uint4* g = (const uint4*)(WrB + (size_t)((kc * 2 + nt) * 32 + lane) * 16);
            wrb[kc][nt].q[0] = g[0]; wrb[kc][nt].q[1] = g[1];
        }
#pragma unroll
    for (int nt = 0; nt < 2; ++nt) {
        const uint4* g = (const uint4*)(WoB + (size_t)(nt * 32 + lane) * 16);
        wob[nt].q[0] = g[0]; wob[nt].q[1] = g[1];
    }

    float bh_l[4], g2_l[4], be2_l[4];
#pragma unroll
    for (int nt = 0; nt < 4; ++nt) {
        bh_l[nt]  = bh[nt * 16 + nlo];
        g2_l[nt]  = ln2g[nt * 16 + nlo];
        be2_l[nt] = ln2b[nt * 16 + nlo];
    }
    float br_l[2] = { br[nlo], br[16 + nlo] };
    float bo_l[2] = { bo[nlo], bo[16 + nlo] };

    for (int i = lane; i < 16 * LAT; i += 32) hbuf[i] = 0.f;
    v8f pool0 = {}, pool1 = {};
    __syncthreads();

#pragma unroll 1
    for (int t = 0; t < T2; ++t) {
        // combined = xt + h (each lane owns one contiguous 16-element run)
        {
            const float* xt = xt_all + ((size_t)t * BB + b0) * LAT + lane * 16;
            const float* hb = hbuf + lane * 16;
            Frag16 cc;
#pragma unroll
            for (int j = 0; j < 16; ++j) cc.h[j] = (_Float16)(xt[j] + hb[j]);
            uint4* cd = (uint4*)(comb16 + lane * 16);
            cd[0] = cc.q[0]; cd[1] = cc.q[1];
        }
        __syncthreads();

        // GEMM1: hp = relu(combined @ Wh^T + bh) -> [16, 64]
        Frag16 a;
        load_a(a, comb16, LAT, nlo, lo);
        float vals[4][8];
#pragma unroll
        for (int nt = 0; nt < 4; ++nt) {
            v8f c = {};
            c = wmma_f16(a.v, whb[nt].v, c);
#pragma unroll
            for (int v = 0; v < 8; ++v) {
                float x = c[v] + bh_l[nt];
                vals[nt][v] = x > 0.f ? x : 0.f;
            }
        }
        // row LayerNorm over HID=64 (butterfly over 16-lane halves)
        float pv[8], qv[8];
#pragma unroll
        for (int v = 0; v < 8; ++v) {
            float s = 0.f, q = 0.f;
#pragma unroll
            for (int nt = 0; nt < 4; ++nt) { s += vals[nt][v]; q += vals[nt][v] * vals[nt][v]; }
            pv[v] = s; qv[v] = q;
        }
#pragma unroll
        for (int mask = 1; mask < 16; mask <<= 1)
#pragma unroll
            for (int v = 0; v < 8; ++v) {
                pv[v] += __shfl_xor(pv[v], mask, 32);
                qv[v] += __shfl_xor(qv[v], mask, 32);
            }
#pragma unroll
        for (int v = 0; v < 8; ++v) {
            float mean = pv[v] * (1.f / HID);
            float var  = qv[v] * (1.f / HID) - mean * mean;
            float r    = rsqrtf(var + EPS_);
            int row = v + rofs;
#pragma unroll
            for (int nt = 0; nt < 4; ++nt) {
                float x = (vals[nt][v] - mean) * r * g2_l[nt] + be2_l[nt];
                hp16[row * HID + nt * 16 + nlo] = (_Float16)x;
            }
        }
        __syncthreads();

        // GEMM2: hnew = tanh(hp_ln @ Wr^T + br) -> [16, 32]
        Frag16 a0, a1;
        load_a(a0, hp16, HID, nlo, lo);
        load_a(a1, hp16 + 32, HID, nlo, lo);
        v8f d0 = {}, d1 = {};
        d0 = wmma_f16(a0.v, wrb[0][0].v, d0);
        d0 = wmma_f16(a1.v, wrb[1][0].v, d0);
        d1 = wmma_f16(a0.v, wrb[0][1].v, d1);
        d1 = wmma_f16(a1.v, wrb[1][1].v, d1);
#pragma unroll
        for (int v = 0; v < 8; ++v) {
            float h0 = tanhf(d0[v] + br_l[0]);
            float h1 = tanhf(d1[v] + br_l[1]);
            pool0[v] += h0;
            pool1[v] += h1;
            int row = v + rofs;
            hbuf[row * LAT + nlo]      = h0;
            hbuf[row * LAT + 16 + nlo] = h1;
        }
        __syncthreads();
    }

    // pooled mean -> final GEMM with Wo
#pragma unroll
    for (int v = 0; v < 8; ++v) {
        int row = v + rofs;
        comb16[row * LAT + nlo]      = (_Float16)(pool0[v] * (1.f / T2));
        comb16[row * LAT + 16 + nlo] = (_Float16)(pool1[v] * (1.f / T2));
    }
    __syncthreads();
    Frag16 pa;
    load_a(pa, comb16, LAT, nlo, lo);
    v8f o0 = {}, o1 = {};
    o0 = wmma_f16(pa.v, wob[0].v, o0);
    o1 = wmma_f16(pa.v, wob[1].v, o1);
#pragma unroll
    for (int v = 0; v < 8; ++v) {
        int row = v + rofs;
        out[(size_t)(b0 + row) * LAT + nlo]      = o0[v] + bo_l[0];
        out[(size_t)(b0 + row) * LAT + 16 + nlo] = o1[v] + bo_l[1];
    }
}

// ================= host launch =================
extern "C" void kernel_launch(void* const* d_in, const int* in_sizes, int n_in,
                              void* d_out, int out_size, void* d_ws, size_t ws_size,
                              hipStream_t stream) {
    const float* x    = (const float*)d_in[0];
    const float* c1w  = (const float*)d_in[1];
    const float* c1b  = (const float*)d_in[2];
    const float* bn1g = (const float*)d_in[3];
    const float* bn1b = (const float*)d_in[4];
    const float* c2w  = (const float*)d_in[5];
    const float* c2b  = (const float*)d_in[6];
    const float* bn2g = (const float*)d_in[7];
    const float* bn2b = (const float*)d_in[8];
    const float* Wi   = (const float*)d_in[9];
    const float* bi   = (const float*)d_in[10];
    const float* ln1g = (const float*)d_in[11];
    const float* ln1b = (const float*)d_in[12];
    const float* Wh   = (const float*)d_in[13];
    const float* bh   = (const float*)d_in[14];
    const float* ln2g = (const float*)d_in[15];
    const float* ln2b = (const float*)d_in[16];
    const float* Wr   = (const float*)d_in[17];
    const float* br   = (const float*)d_in[18];
    const float* Wo   = (const float*)d_in[19];
    const float* bo   = (const float*)d_in[20];

    char* ws = (char*)d_ws;
    _Float16* A1  = (_Float16*)(ws + OFF_A1);
    _Float16* A2  = (_Float16*)(ws + OFF_A2);
    _Float16* WhB = (_Float16*)(ws + OFF_WHB);
    _Float16* WrB = (_Float16*)(ws + OFF_WRB);
    _Float16* WoB = (_Float16*)(ws + OFF_WOB);
    _Float16* WiA = (_Float16*)(ws + OFF_WIA);
    float* stats  = (float*)(ws + OFF_STATS);
    float *sum1 = stats,       *sq1 = stats + 32;
    float *sum2 = stats + 64,  *sq2 = stats + 96;
    float *sc1  = stats + 128, *sh1 = stats + 160;
    float *sc2  = stats + 192, *sh2 = stats + 224;
    _Float16* y1t = (_Float16*)(ws + OFF_Y1);
    _Float16* y2t = (_Float16*)(ws + OFF_Y2);
    float*    xt  = (float*)(ws + OFF_XT);

    zero_f32<<<1, 256, 0, stream>>>(stats, 128);
    pack_a<<<(2 * NK1 * 512 + 255) / 256, 256, 0, stream>>>(c1w, A1, K1TOT, NK1);
    pack_a<<<(2 * NK2 * 512 + 255) / 256, 256, 0, stream>>>(c2w, A2, K2TOT, NK2);
    pack_a<<<(2 * 512 + 255) / 256, 256, 0, stream>>>(Wi, WiA, LAT, 1);
    pack_b<<<(4 * 512 + 255) / 256, 256, 0, stream>>>(Wh, WhB, LAT, HID);
    pack_b<<<(4 * 512 + 255) / 256, 256, 0, stream>>>(Wr, WrB, HID, LAT);
    pack_b<<<(2 * 512 + 255) / 256, 256, 0, stream>>>(Wo, WoB, LAT, LAT);

    conv1_kernel<<<dim3(T1 / 16, BB), 64, 0, stream>>>(x, A1, c1b, y1t, sum1, sq1);
    bn_params<<<1, 32, 0, stream>>>(sum1, sq1, bn1g, bn1b, sc1, sh1,
                                    1.f / ((float)BB * (float)T1));
    conv2_kernel<<<dim3((T2 + 15) / 16, BB), 64, 0, stream>>>(y1t, A2, c2b, sc1, sh1,
                                                              y2t, sum2, sq2);
    bn_params<<<1, 32, 0, stream>>>(sum2, sq2, bn2g, bn2b, sc2, sh2,
                                    1.f / ((float)BB * (float)T2));
    proj_kernel<<<(T2 * BB) / 32, 64, 0, stream>>>(y2t, sc2, sh2, WiA, bi,
                                                   ln1g, ln1b, xt);
    rnn_kernel<<<BB / 16, 32, 0, stream>>>(xt, WhB, bh, ln2g, ln2b, WrB, br, WoB, bo,
                                           (float*)d_out);
}
